// MLP_61220463837807
// MI455X (gfx1250) — 16-bit and 8-bit matrix operand order repaired and checked by reading, compiled, not run on hardware
//
#include <hip/hip_runtime.h>


#define M_DIM 8192
#define D_DIM 2048
#define H_DIM 8192
#define EPSQ  1e-5f

typedef _Float16 v16h __attribute__((ext_vector_type(16)));
typedef _Float16 v8h  __attribute__((ext_vector_type(8)));
typedef _Float16 v4h  __attribute__((ext_vector_type(4)));
typedef float    v8f  __attribute__((ext_vector_type(8)));
typedef int      v8i  __attribute__((ext_vector_type(8)));
typedef int      v4i  __attribute__((ext_vector_type(4)));
typedef int      v2i  __attribute__((ext_vector_type(2)));

__device__ __forceinline__ float half16_max(float v) {
    v = fmaxf(v, __int_as_float(__builtin_amdgcn_ds_swizzle(__float_as_int(v), 0x041F)));
    v = fmaxf(v, __int_as_float(__builtin_amdgcn_ds_swizzle(__float_as_int(v), 0x081F)));
    v = fmaxf(v, __int_as_float(__builtin_amdgcn_ds_swizzle(__float_as_int(v), 0x101F)));
    v = fmaxf(v, __int_as_float(__builtin_amdgcn_ds_swizzle(__float_as_int(v), 0x201F)));
    return v;
}


__global__ void init_ws_kernel(float* sums, unsigned* rowmax) {
    int i = blockIdx.x * blockDim.x + threadIdx.x;
    if (i < 2) sums[i] = 0.0f;
    if (i < M_DIM) rowmax[i] = 0u;
}

__global__ void absum_kernel(const float* __restrict__ w, float* __restrict__ acc, int n4) {
    __shared__ float sdata[256];
    float s = 0.0f;
    int stride = gridDim.x * blockDim.x;
    for (int i = blockIdx.x * blockDim.x + threadIdx.x; i < n4; i += stride) {
        float4 v = reinterpret_cast<const float4*>(w)[i];
        s += fabsf(v.x) + fabsf(v.y) + fabsf(v.z) + fabsf(v.w);
    }
    sdata[threadIdx.x] = s;
    __syncthreads();
    for (int off = 128; off > 0; off >>= 1) {
        if ((int)threadIdx.x < off) sdata[threadIdx.x] += sdata[threadIdx.x + off];
        __syncthreads();
    }
    if (threadIdx.x == 0) atomicAdd(acc, sdata[0]);
}

__global__ void cvt_x_kernel(const float* __restrict__ x, _Float16* __restrict__ xh, int n4) {
    int stride = gridDim.x * blockDim.x;
    for (int i = blockIdx.x * blockDim.x + threadIdx.x; i < n4; i += stride) {
        float4 v = reinterpret_cast<const float4*>(x)[i];
        v4h o;
        o[0] = (_Float16)v.x; o[1] = (_Float16)v.y;
        o[2] = (_Float16)v.z; o[3] = (_Float16)v.w;
        reinterpret_cast<v4h*>(xh)[i] = o;
    }
}

__global__ void quant_w1_kernel(const float* __restrict__ cfc,
                                const float* __restrict__ sums,
                                _Float16* __restrict__ w1q) {
    __shared__ float tile[32][33];
    float scale = fmaxf(sums[0] * (1.0f / (float)(D_DIM * (long)H_DIM)), EPSQ);
    int h0 = blockIdx.x * 32, d0 = blockIdx.y * 32;
    for (int r = threadIdx.y; r < 32; r += 8) {
        float v = cfc[(size_t)(d0 + r) * H_DIM + h0 + threadIdx.x];
        tile[r][threadIdx.x] = fminf(fmaxf(rintf(v / scale), -1.0f), 1.0f);
    }
    __syncthreads();
    for (int r = threadIdx.y; r < 32; r += 8) {
        w1q[(size_t)(h0 + r) * D_DIM + d0 + threadIdx.x] = (_Float16)tile[threadIdx.x][r];
    }
}

__global__ void quant_w2_kernel(const float* __restrict__ cproj,
                                const float* __restrict__ sums,
                                char* __restrict__ q2, int n4) {
    float scale = fmaxf(sums[1] * (1.0f / (float)(D_DIM * (long)H_DIM)), EPSQ);
    int stride = gridDim.x * blockDim.x;
    for (int i = blockIdx.x * blockDim.x + threadIdx.x; i < n4; i += stride) {
        float4 v = reinterpret_cast<const float4*>(cproj)[i];
        char4 o;
        o.x = (char)(int)fminf(fmaxf(rintf(v.x / scale), -1.0f), 1.0f);
        o.y = (char)(int)fminf(fmaxf(rintf(v.y / scale), -1.0f), 1.0f);
        o.z = (char)(int)fminf(fmaxf(rintf(v.z / scale), -1.0f), 1.0f);
        o.w = (char)(int)fminf(fmaxf(rintf(v.w / scale), -1.0f), 1.0f);
        reinterpret_cast<char4*>(q2)[i] = o;
    }
}

__global__ __launch_bounds__(256) void gemm1_kernel(
    const _Float16* __restrict__ xh,
    const _Float16* __restrict__ w1q,
    const float* __restrict__ sums,
    _Float16* __restrict__ hsq,
    unsigned* __restrict__ rowmax) {
    const int lane = threadIdx.x & 31;
    const int wid  = threadIdx.x >> 5;
    const int wn = wid & 3, wm = wid >> 2;
    const int half = lane >> 4, l16 = lane & 15;
    const int baseM = blockIdx.y * 128 + wm * 64;
    const int baseN = blockIdx.x * 256 + wn * 64;

    v8f acc[4][4] = {};

    const _Float16* aptr[4];
#pragma unroll
    for (int t = 0; t < 4; ++t)
        aptr[t] = xh + (size_t)(baseM + t * 16 + l16) * D_DIM + half * 16;
    const _Float16* bptr[4];
#pragma unroll
    for (int t = 0; t < 4; ++t)
        bptr[t] = w1q + (size_t)(baseN + t * 16 + l16) * D_DIM + half * 16;

    for (int kk = 0; kk < D_DIM; kk += 32) {
        v16h afrag[4], bfrag[4];
#pragma unroll
        for (int t = 0; t < 4; ++t) {
            v8h lo = *reinterpret_cast<const v8h*>(aptr[t] + kk);
            v8h hi = *reinterpret_cast<const v8h*>(aptr[t] + kk + 8);
#pragma unroll
            for (int i = 0; i < 8; ++i) { afrag[t][i] = lo[i]; afrag[t][i + 8] = hi[i]; }
        }
#pragma unroll
        for (int t = 0; t < 4; ++t)
            bfrag[t] = *reinterpret_cast<const v16h*>(bptr[t] + kk);
#pragma unroll
        for (int tm = 0; tm < 4; ++tm)
#pragma unroll
            for (int tn = 0; tn < 4; ++tn)
                acc[tm][tn] = __builtin_amdgcn_wmma_f32_16x16x32_f16(
                    false, afrag[tm], false, bfrag[tn], (short)0, acc[tm][tn], false, false);
    }

    const float scale1 = fmaxf(sums[0] * (1.0f / (float)(D_DIM * (long)H_DIM)), EPSQ);
#pragma unroll
    for (int tm = 0; tm < 4; ++tm) {
#pragma unroll
        for (int v = 0; v < 8; ++v) {
            const int mG = baseM + tm * 16 + v + 8 * half;
            float rmax = 0.0f;
#pragma unroll
            for (int tn = 0; tn < 4; ++tn) {
                const int nG = baseN + tn * 16 + l16;
                float hval = scale1 * acc[tm][tn][v];
                float r = fmaxf(hval, 0.0f);
                _Float16 sq = (_Float16)(r * r);
                hsq[(size_t)mG * H_DIM + nG] = sq;
                rmax = fmaxf(rmax, (float)sq);
            }
            rmax = half16_max(rmax);
            if (l16 == 0) atomicMax(rowmax + mG, __float_as_uint(rmax));
        }
    }
}

__global__ void quant_h_kernel(const _Float16* __restrict__ hsq,
                               const unsigned* __restrict__ rowmax,
                               const float* __restrict__ sums,
                               unsigned char* __restrict__ hq,
                               float* __restrict__ rowfac) {
    const int m = blockIdx.y;
    const int col = blockIdx.x * 256 + threadIdx.x;
    const float xmax = fmaxf(__uint_as_float(rowmax[m]), EPSQ);
    const float inv = 255.0f / xmax;
    float v = (float)hsq[(size_t)m * H_DIM + col];
    float q = rintf(v * inv);
    hq[(size_t)m * H_DIM + col] = (unsigned char)(int)fminf(fmaxf(q, 0.0f), 255.0f);
    if (col == 0) {
        float scale2 = fmaxf(sums[1] * (1.0f / (float)(D_DIM * (long)H_DIM)), EPSQ);
        rowfac[m] = xmax * scale2 * (1.0f / 255.0f);
    }
}

__global__ __launch_bounds__(256) void gemm2_kernel(
    const unsigned char* __restrict__ hq,
    const char* __restrict__ q2,
    const float* __restrict__ rowfac,
    float* __restrict__ out) {
    const int lane = threadIdx.x & 31;
    const int wid  = threadIdx.x >> 5;
    const int wn = wid & 3, wm = wid >> 2;
    const int half = lane >> 4, l16 = lane & 15;
    const int baseM = blockIdx.y * 128 + wm * 64;
    const int baseN = blockIdx.x * 256 + wn * 64;

    v8i acc[4][4] = {};

    const unsigned char* aptr[4];
#pragma unroll
    for (int t = 0; t < 4; ++t)
        aptr[t] = hq + (size_t)(baseM + t * 16 + l16) * H_DIM + half * 16;
    const char* bptr[4];
#pragma unroll
    for (int t = 0; t < 4; ++t)
        bptr[t] = q2 + (size_t)(baseN + t * 16 + l16) * H_DIM + half * 16;

    for (int kk = 0; kk < H_DIM; kk += 64) {
        v8i afrag[4], bfrag[4];
#pragma unroll
        for (int t = 0; t < 4; ++t) {
            v2i c0 = *reinterpret_cast<const v2i*>(aptr[t] + kk);
            v2i c1 = *reinterpret_cast<const v2i*>(aptr[t] + kk + 8);
            v2i c2 = *reinterpret_cast<const v2i*>(aptr[t] + kk + 32);
            v2i c3 = *reinterpret_cast<const v2i*>(aptr[t] + kk + 40);
            v8i a;
            a[0] = c0[0]; a[1] = c0[1]; a[2] = c1[0]; a[3] = c1[1];
            a[4] = c2[0]; a[5] = c2[1]; a[6] = c3[0]; a[7] = c3[1];
            afrag[t] = a;
        }
#pragma unroll
        for (int t = 0; t < 4; ++t) {
            v4i d0 = *reinterpret_cast<const v4i*>(bptr[t] + kk);
            v4i d1 = *reinterpret_cast<const v4i*>(bptr[t] + kk + 32);
            v8i b;
            b[0] = d0[0]; b[1] = d0[1]; b[2] = d0[2]; b[3] = d0[3];
            b[4] = d1[0]; b[5] = d1[1]; b[6] = d1[2]; b[7] = d1[3];
            bfrag[t] = b;
        }
#pragma unroll
        for (int tm = 0; tm < 4; ++tm)
#pragma unroll
            for (int tn = 0; tn < 4; ++tn)
                acc[tm][tn] = __builtin_amdgcn_wmma_i32_16x16x64_iu8(
                    false  , afrag[tm], true  , bfrag[tn],
                    acc[tm][tn], false, false);
    }

#pragma unroll
    for (int tm = 0; tm < 4; ++tm) {
#pragma unroll
        for (int v = 0; v < 8; ++v) {
            const int mG = baseM + tm * 16 + v + 8 * half;
            const float f = rowfac[mG];
#pragma unroll
            for (int tn = 0; tn < 4; ++tn) {
                const int nG = baseN + tn * 16 + l16;
                out[(size_t)mG * D_DIM + nG] = (float)acc[tm][tn][v] * f;
            }
        }
    }
}


extern "C" void kernel_launch(void* const* d_in, const int* in_sizes, int n_in,
                              void* d_out, int out_size, void* d_ws, size_t ws_size,
                              hipStream_t stream) {
    (void)in_sizes; (void)n_in; (void)out_size; (void)ws_size;
    const float* x     = (const float*)d_in[0];
    const float* cfc   = (const float*)d_in[1];
    const float* cproj = (const float*)d_in[2];
    float* out = (float*)d_out;

    char* ws = (char*)d_ws;
    float*    sums   = (float*)ws;
    unsigned* rowmax = (unsigned*)(ws + 256);
    float*    rowfac = (float*)(ws + 256 + (size_t)M_DIM * 4);
    size_t off = 256 + (size_t)M_DIM * 8;
    _Float16* xh  = (_Float16*)(ws + off); off += (size_t)M_DIM * D_DIM * 2;
    _Float16* w1q = (_Float16*)(ws + off); off += (size_t)H_DIM * D_DIM * 2;
    char*     q2  = (char*)(ws + off);     off += (size_t)D_DIM * H_DIM;
    _Float16* hsq = (_Float16*)(ws + off); off += (size_t)M_DIM * H_DIM * 2;
    unsigned char* hq = (unsigned char*)(ws + off);

    const int NW = D_DIM * H_DIM;
    const int ND = M_DIM * D_DIM;

    init_ws_kernel<<<(M_DIM + 255) / 256, 256, 0, stream>>>(sums, rowmax);
    absum_kernel<<<2048, 256, 0, stream>>>(cfc,   sums + 0, NW / 4);
    absum_kernel<<<2048, 256, 0, stream>>>(cproj, sums + 1, NW / 4);
    cvt_x_kernel<<<4096, 256, 0, stream>>>(x, xh, ND / 4);
    quant_w1_kernel<<<dim3(H_DIM / 32, D_DIM / 32), dim3(32, 8), 0, stream>>>(cfc, sums, w1q);
    quant_w2_kernel<<<4096, 256, 0, stream>>>(cproj, sums, q2, NW / 4);
    gemm1_kernel<<<dim3(H_DIM / 256, M_DIM / 128), 256, 0, stream>>>(xh, w1q, sums, hsq, rowmax);
    quant_h_kernel<<<dim3(H_DIM / 256, M_DIM), 256, 0, stream>>>(hsq, rowmax, sums, hq, rowfac);
    gemm2_kernel<<<dim3(D_DIM / 256, M_DIM / 128), 256, 0, stream>>>(hq, q2, rowfac, out);
}
